// CategoryBranch_39067022525173
// MI455X (gfx1250) — matrix operand order repaired and checked by reading, not compiled, not run on hardware
//
#include <hip/hip_runtime.h>

#define BB 32
#define TIN 1024
#define TOUT 256
#define DIN 128
#define HH 512
#define KK 5
#define CC 10
#define H3 1536

typedef __attribute__((ext_vector_type(16))) _Float16 v16h;
typedef __attribute__((ext_vector_type(8)))  _Float16 v8h;
typedef __attribute__((ext_vector_type(8)))  float    v8f;

// ---------------------------------------------------------------------------
// WMMA helpers
// ---------------------------------------------------------------------------
__device__ __forceinline__ v8f wmma_f16(v16h a, v16h b, v8f c) {
  // D = A(16x32 f16) * B(32x16 f16) + C(16x16 f32)
  return __builtin_amdgcn_wmma_f32_16x16x32_f16(
      /*neg_a=*/false, a, /*neg_b=*/false, b,
      /*c_mod=*/(short)0, c, /*reuse_a=*/false, /*reuse_b=*/false);
}

// Load one 16x32 f16 A-tile per the CDNA5 striped layout:
// lane<16: rows m0..m0+15, K = k0 + {0..7, 16..23}
// lane>=16: same rows,     K = k0 + {8..15, 24..31}
__device__ __forceinline__ v16h load_a_tile(const _Float16* __restrict__ A,
                                            int ldk, int m0, int k0, int lane) {
  int m  = m0 + (lane & 15);
  int kb = (lane < 16) ? 0 : 8;
  const _Float16* p = A + (size_t)m * ldk + k0 + kb;
  v8h lo = *(const v8h*)(p);        // K = kb .. kb+7   (16B aligned)
  v8h hi = *(const v8h*)(p + 16);   // K = kb+16 .. kb+23
  v16h r;
#pragma unroll
  for (int i = 0; i < 8; ++i) { r[i] = lo[i]; r[8 + i] = hi[i]; }
  return r;
}

__device__ __forceinline__ v16h load_b_tile(const _Float16* __restrict__ Bpk,
                                            int tile, int lane) {
  return *(const v16h*)(Bpk + ((size_t)tile * 32 + lane) * 16);
}

__device__ __forceinline__ float sigmoidf_(float x) {
  return 1.0f / (1.0f + __expf(-x));
}

// ---------------------------------------------------------------------------
// Prep kernels
// ---------------------------------------------------------------------------
__global__ void cvt_f32_to_f16(const float* __restrict__ src,
                               _Float16* __restrict__ dst, int n) {
  int i = blockIdx.x * 256 + threadIdx.x;
  if (i < n) dst[i] = (_Float16)src[i];
}

// Pack a KxN fp32 row-major matrix into WMMA B-operand tiles (32x16 per tile).
__global__ void pack_b_tiles(const float* __restrict__ W, _Float16* __restrict__ pk,
                             int ldN, int ntn, int total) {
  int p = blockIdx.x * 256 + threadIdx.x;
  if (p >= total) return;
  int v    = p & 15;
  int lane = (p >> 4) & 31;
  int tile = p >> 9;
  int tk = tile / ntn, tn = tile % ntn;
  int k = tk * 32 + ((lane & 16) ? 8 : 0) + (v < 8 ? v : v + 8);
  int n = tn * 16 + (lane & 15);
  pk[p] = (_Float16)W[(size_t)k * ldN + n];
}

__global__ void init_state(float* h32a, float* h32b,
                           _Float16* h16a, _Float16* h16b, float* kappa) {
  int i = blockIdx.x * 256 + threadIdx.x;
  if (i < BB * HH) {
    h32a[i] = 0.f; h32b[i] = 0.f;
    h16a[i] = (_Float16)0.f; h16b[i] = (_Float16)0.f;
  }
  if (i < BB * KK) kappa[i] = 0.f;
}

// ---------------------------------------------------------------------------
// Encoder: context_f16 = f16( input_f16 @ W_enc + b_enc )
// grid (M/16 = 2048, N/64 = 8), one wave per block; occupancy hides latency.
// ---------------------------------------------------------------------------
__global__ __launch_bounds__(32)
void encoder_wmma(const _Float16* __restrict__ Af, const _Float16* __restrict__ Bpk,
                  const float* __restrict__ bias, _Float16* __restrict__ ctx) {
  int lane = threadIdx.x;
  int m0 = blockIdx.x * 16;
  int n0 = blockIdx.y * 64;
  int tnb = n0 >> 4;
  v8f acc[4] = {};
#pragma unroll
  for (int tk = 0; tk < 4; ++tk) {
    v16h a = load_a_tile(Af, DIN, m0, tk * 32, lane);
#pragma unroll
    for (int j = 0; j < 4; ++j) {
      v16h b = load_b_tile(Bpk, tk * 32 + tnb + j, lane);
      acc[j] = wmma_f16(a, b, acc[j]);
    }
  }
  int nbase = n0 + (lane & 15);
  int mbase = m0 + ((lane >> 4) << 3);
#pragma unroll
  for (int j = 0; j < 4; ++j) {
    int n = nbase + j * 16;
    float bv = bias[n];
#pragma unroll
    for (int i = 0; i < 8; ++i)
      ctx[(size_t)(mbase + i) * HH + n] = (_Float16)(acc[j][i] + bv);
  }
}

// ---------------------------------------------------------------------------
// Fused step kernel: one block per batch.
//   phase 0: logits(step-1) dots + W_att dots in disjoint thread ranges
//   phase 1: alpha/beta/kappa finalize + logits write
//   phase 2: phi -> LDS (+ out_weights)
//   phase 3: attended sweep over L2-resident f16 context (b128 + prefetch)
// ---------------------------------------------------------------------------
__global__ __launch_bounds__(256)
void attn_step(const float* __restrict__ h32, const float* __restrict__ Watt,
               const float* __restrict__ Wout, const float* __restrict__ bout,
               float* __restrict__ kappa, const _Float16* __restrict__ ctx,
               _Float16* __restrict__ att16, float* __restrict__ out_w,
               float* __restrict__ out_h, int step) {
  __shared__ float sph[TIN];
  __shared__ float part[4][HH];
  __shared__ float redA[3 * KK][8];
  __shared__ float redL[CC][8];
  __shared__ float alpha[KK], beta[KK], kap[KK];
  int b = blockIdx.x, tid = threadIdx.x;
  const float* hr = h32 + b * HH;

  // phase 0: 120 threads do the 15 W_att dots, 80 threads do the 10 W_out dots
  if (tid < 8 * 3 * KK) {
    int col = tid >> 3, r = tid & 7;
    float s = 0.f;
    for (int h = r; h < HH; h += 8) s += hr[h] * Watt[h * (3 * KK) + col];
    redA[col][r] = s;
  } else if (tid < 8 * 3 * KK + 8 * CC) {
    int idx = tid - 8 * 3 * KK;
    int c = idx >> 3, r = idx & 7;
    float s = 0.f;
    for (int h = r; h < HH; h += 8) s += hr[h] * Wout[h * CC + c];
    redL[c][r] = s;
  }
  __syncthreads();

  // phase 1
  if (tid < 3 * KK) {
    float s = 0.f;
#pragma unroll
    for (int r = 0; r < 8; ++r) s += redA[tid][r];
    float p = __expf(s);
    if (tid < KK)          alpha[tid] = p;
    else if (tid < 2 * KK) beta[tid - KK] = p;
    else {
      int k = tid - 2 * KK;
      float kn = kappa[b * KK + k] + p;   // kappa_new = kappa + dkappa
      kappa[b * KK + k] = kn;
      kap[k] = kn;
    }
  } else if (tid >= 32 && tid < 32 + CC && step > 0) {
    int c = tid - 32;                     // logits of PREVIOUS step
    float s = bout[c];
#pragma unroll
    for (int r = 0; r < 8; ++r) s += redL[c][r];
    out_h[((size_t)b * TOUT + (step - 1)) * CC + c] = s;
  }
  __syncthreads();

  // phase 2: phi (Gaussian mixture over encoder positions)
  for (int t = tid; t < TIN; t += 256) {
    float ph = 0.f;
#pragma unroll
    for (int k = 0; k < KK; ++k) {
      float d = kap[k] - (float)t;
      ph += alpha[k] * __expf(-beta[k] * d * d);
    }
    sph[t] = ph;
    out_w[((size_t)b * TOUT + step) * TIN + t] = ph;
  }
  __syncthreads();

  // phase 3: attended[b,h] = sum_t phi[t] * ctx[b,t,h]
  int g  = tid >> 6;                 // t-group 0..3
  int c  = tid & 63;                 // column octet 0..63
  int h0 = c * 8;
  float acc[8] = {0.f, 0.f, 0.f, 0.f, 0.f, 0.f, 0.f, 0.f};
  const _Float16* base = ctx + (size_t)b * TIN * HH + h0;
  for (int t = g; t < TIN; t += 4) {
    __builtin_prefetch((const void*)(base + (size_t)(t + 16) * HH), 0, 0);
    v8h v = *(const v8h*)(base + (size_t)t * HH);
    float ph = sph[t];
#pragma unroll
    for (int i = 0; i < 8; ++i) acc[i] += ph * (float)v[i];
  }
#pragma unroll
  for (int i = 0; i < 8; ++i) part[g][h0 + i] = acc[i];
  __syncthreads();
  int h = tid * 2;
#pragma unroll
  for (int j = 0; j < 2; ++j) {
    float s = part[0][h + j] + part[1][h + j] + part[2][h + j] + part[3][h + j];
    att16[b * HH + h + j] = (_Float16)s;
  }
}

// ---------------------------------------------------------------------------
// Step kernel: GRU cell via WMMA, software-pipelined k-loop.
// One wave per 16-column group (32 blocks); loads for k+1 issued before the
// 12 WMMAs of k; gate math done in C-layout registers.
// ---------------------------------------------------------------------------
__global__ __launch_bounds__(32)
void gru_wmma(const _Float16* __restrict__ att16, const _Float16* __restrict__ h16cur,
              const float* __restrict__ h32cur,
              const _Float16* __restrict__ Wih, const _Float16* __restrict__ Whh,
              float* __restrict__ h32nxt, _Float16* __restrict__ h16nxt) {
  int lane = threadIdx.x;
  int blk  = blockIdx.x;                 // column group: n in [blk*16, blk*16+16)
  v8f ai[3][2] = {};                     // gate tiles from attended @ W_ih
  v8f ah[3][2] = {};                     // gate tiles from hidden   @ W_hh
  v16h ax0 = load_a_tile(att16,  HH, 0,  0, lane);
  v16h ax1 = load_a_tile(att16,  HH, 16, 0, lane);
  v16h ad0 = load_a_tile(h16cur, HH, 0,  0, lane);
  v16h ad1 = load_a_tile(h16cur, HH, 16, 0, lane);
  v16h bi[3], bh[3];
#pragma unroll
  for (int gg = 0; gg < 3; ++gg) {
    bi[gg] = load_b_tile(Wih, gg * 32 + blk, lane);
    bh[gg] = load_b_tile(Whh, gg * 32 + blk, lane);
  }
  for (int tk = 0; tk < 16; ++tk) {
    v16h nax0, nax1, nad0, nad1, nbi[3], nbh[3];
    if (tk < 15) {
      int k0 = (tk + 1) * 32;
      nax0 = load_a_tile(att16,  HH, 0,  k0, lane);
      nax1 = load_a_tile(att16,  HH, 16, k0, lane);
      nad0 = load_a_tile(h16cur, HH, 0,  k0, lane);
      nad1 = load_a_tile(h16cur, HH, 16, k0, lane);
#pragma unroll
      for (int gg = 0; gg < 3; ++gg) {
        int tn = (tk + 1) * 96 + gg * 32 + blk;
        nbi[gg] = load_b_tile(Wih, tn, lane);
        nbh[gg] = load_b_tile(Whh, tn, lane);
      }
    }
#pragma unroll
    for (int gg = 0; gg < 3; ++gg) {     // r, z, n gates
      ai[gg][0] = wmma_f16(ax0, bi[gg], ai[gg][0]);
      ai[gg][1] = wmma_f16(ax1, bi[gg], ai[gg][1]);
      ah[gg][0] = wmma_f16(ad0, bh[gg], ah[gg][0]);
      ah[gg][1] = wmma_f16(ad1, bh[gg], ah[gg][1]);
    }
    if (tk < 15) {
      ax0 = nax0; ax1 = nax1; ad0 = nad0; ad1 = nad1;
#pragma unroll
      for (int gg = 0; gg < 3; ++gg) { bi[gg] = nbi[gg]; bh[gg] = nbh[gg]; }
    }
  }
  int n = blk * 16 + (lane & 15);
#pragma unroll
  for (int mt = 0; mt < 2; ++mt) {
    int mbase = mt * 16 + ((lane >> 4) << 3);
#pragma unroll
    for (int i = 0; i < 8; ++i) {
      int m = mbase + i;                 // batch row
      float hprev = h32cur[m * HH + n];
      float r  = sigmoidf_(ai[0][mt][i] + ah[0][mt][i]);
      float z  = sigmoidf_(ai[1][mt][i] + ah[1][mt][i]);
      float nn = tanhf(ai[2][mt][i] + r * ah[2][mt][i]);
      float hnew = (1.f - z) * nn + z * hprev;
      h32nxt[m * HH + n] = hnew;
      h16nxt[m * HH + n] = (_Float16)hnew;
    }
  }
}

// ---------------------------------------------------------------------------
// Final logits for the last step (all earlier steps are fused into attn_step).
// ---------------------------------------------------------------------------
__global__ __launch_bounds__(320)
void logits_k(const float* __restrict__ h32, const float* __restrict__ Wout,
              const float* __restrict__ bout, float* __restrict__ out_h, int step) {
  int tid = threadIdx.x;
  if (tid >= BB * CC) return;
  int b = tid / CC, c = tid % CC;
  const float* hr = h32 + b * HH;
  float s = bout[c];
  for (int h = 0; h < HH; ++h) s += hr[h] * Wout[h * CC + c];
  out_h[((size_t)b * TOUT + step) * CC + c] = s;
}

// ---------------------------------------------------------------------------
// Host
// ---------------------------------------------------------------------------
extern "C" void kernel_launch(void* const* d_in, const int* in_sizes, int n_in,
                              void* d_out, int out_size, void* d_ws, size_t ws_size,
                              hipStream_t stream) {
  (void)in_sizes; (void)n_in; (void)out_size; (void)ws_size;
  const float* input_ = (const float*)d_in[0];
  /* d_in[1] "output" (int labels) unused by the forward math */
  const float* W_enc  = (const float*)d_in[2];
  const float* b_enc  = (const float*)d_in[3];
  const float* W_att  = (const float*)d_in[4];
  const float* W_ih   = (const float*)d_in[5];
  const float* W_hh   = (const float*)d_in[6];
  const float* W_out  = (const float*)d_in[7];
  const float* b_out  = (const float*)d_in[8];

  float* out_hidden  = (float*)d_out;                          // (B, TOUT, C)
  float* out_weights = (float*)d_out + (size_t)BB * TOUT * CC; // (B, TOUT, TIN)

  char* w = (char*)d_ws;
  auto carve = [&](size_t bytes) -> char* {
    char* p = w; w += (bytes + 255) & ~(size_t)255; return p;
  };
  _Float16* in16    = (_Float16*)carve((size_t)BB * TIN * DIN * 2);
  _Float16* ctx16   = (_Float16*)carve((size_t)BB * TIN * HH * 2);
  _Float16* wenc_pk = (_Float16*)carve((size_t)DIN * HH * 2);
  _Float16* wih_pk  = (_Float16*)carve((size_t)HH * H3 * 2);
  _Float16* whh_pk  = (_Float16*)carve((size_t)HH * H3 * 2);
  float*    h32a    = (float*)   carve((size_t)BB * HH * 4);
  float*    h32b    = (float*)   carve((size_t)BB * HH * 4);
  _Float16* h16a    = (_Float16*)carve((size_t)BB * HH * 2);
  _Float16* h16b    = (_Float16*)carve((size_t)BB * HH * 2);
  _Float16* att16   = (_Float16*)carve((size_t)BB * HH * 2);
  float*    kappa   = (float*)   carve((size_t)BB * KK * 4);

  // ---- one-time per call: precision conversion, weight packing, encoder ----
  {
    int n = BB * TIN * DIN;
    cvt_f32_to_f16<<<(n + 255) / 256, 256, 0, stream>>>(input_, in16, n);
  }
  pack_b_tiles<<<(DIN * HH + 255) / 256, 256, 0, stream>>>(W_enc, wenc_pk, HH, HH / 16, DIN * HH);
  pack_b_tiles<<<(HH * H3 + 255) / 256, 256, 0, stream>>>(W_ih, wih_pk, H3, H3 / 16, HH * H3);
  pack_b_tiles<<<(HH * H3 + 255) / 256, 256, 0, stream>>>(W_hh, whh_pk, H3, H3 / 16, HH * H3);
  init_state<<<(BB * HH + 255) / 256, 256, 0, stream>>>(h32a, h32b, h16a, h16b, kappa);
  encoder_wmma<<<dim3(BB * TIN / 16, HH / 64), 32, 0, stream>>>(in16, wenc_pk, b_enc, ctx16);

  // ---- sequential scan: 256 steps, 2 kernels each, double-buffered hidden --
  float*    h32p[2] = { h32a, h32b };
  _Float16* h16p[2] = { h16a, h16b };
  int cur = 0;
  for (int step = 0; step < TOUT; ++step) {
    attn_step<<<BB, 256, 0, stream>>>(h32p[cur], W_att, W_out, b_out, kappa,
                                      ctx16, att16, out_weights, out_hidden, step);
    gru_wmma<<<HH / 16, 32, 0, stream>>>(att16, h16p[cur], h32p[cur],
                                         wih_pk, whh_pk, h32p[cur ^ 1], h16p[cur ^ 1]);
    cur ^= 1;
  }
  logits_k<<<1, 320, 0, stream>>>(h32p[cur], W_out, b_out, out_hidden, TOUT - 1);
}
